// LightMutilHeadSelfAttention_17557826306763
// MI455X (gfx1250) — matrix operand order repaired and checked by reading, not compiled, not run on hardware
//
#include <hip/hip_runtime.h>
#include <hip/hip_bf16.h>

// ---------------------------------------------------------------------------
// Problem constants (from reference): B=8, N=1024, C=768, NH=12, HD=64
// ---------------------------------------------------------------------------
#define BB 8
#define NN 1024
#define CC 768
#define NH 12
#define HD 64

typedef __attribute__((ext_vector_type(16))) _Float16 v16h;
typedef __attribute__((ext_vector_type(8)))  _Float16 v8h;
typedef __attribute__((ext_vector_type(8)))  float    v8f;

__device__ __forceinline__ int lane_id() { return threadIdx.x & 31; }

// ---------------------------------------------------------------------------
// WMMA fragment loaders (CDNA5 wave32 layouts, CDNA5 ISA §7.12.2)
//
// A (16x32 f16, MxK): lane L holds row M=L%16;
//   halfs h=0..7  -> K = (L/16)*8 + h
//   halfs h=8..15 -> K = 16 + (L/16)*8 + (h-8)
// => two aligned 16-byte loads from row-major [M][K] storage.
//
// B (32x16 f16, KxN): lane L holds col N=L%16; same K order as A
// => two aligned 16-byte loads from [N][K] storage (i.e. "B transposed").
// ---------------------------------------------------------------------------
__device__ __forceinline__ v16h load_a_frag(const _Float16* base, int ld,
                                            int row0, int k0) {
    const int l = lane_id();
    const _Float16* p = base + (size_t)(row0 + (l & 15)) * ld + k0 + (l >> 4) * 8;
    v8h x0 = *(const v8h*)(p);
    v8h x1 = *(const v8h*)(p + 16);
    return __builtin_shufflevector(x0, x1,
        0, 1, 2, 3, 4, 5, 6, 7, 8, 9, 10, 11, 12, 13, 14, 15);
}

__device__ __forceinline__ v16h load_b_frag(const _Float16* base, int ld,
                                            int n0, int k0) {
    const int l = lane_id();
    const _Float16* p = base + (size_t)(n0 + (l & 15)) * ld + k0 + (l >> 4) * 8;
    v8h x0 = *(const v8h*)(p);
    v8h x1 = *(const v8h*)(p + 16);
    return __builtin_shufflevector(x0, x1,
        0, 1, 2, 3, 4, 5, 6, 7, 8, 9, 10, 11, 12, 13, 14, 15);
}

__device__ __forceinline__ v8f wmma_f16(v16h a, v16h b, v8f c) {
    return __builtin_amdgcn_wmma_f32_16x16x32_f16(
        /*neg_a=*/false, a, /*neg_b=*/false, b,
        /*c_mod=*/(short)0, c, /*reuse_a=*/false, /*reuse_b=*/false);
}

// ---------------------------------------------------------------------------
// fp32 -> f16 conversion (x, Wq, Wkv)
// ---------------------------------------------------------------------------
__global__ void cvt_f32_to_f16(const float* __restrict__ src,
                               _Float16* __restrict__ dst, int n) {
    int i = blockIdx.x * blockDim.x + threadIdx.x;
    int stride = gridDim.x * blockDim.x;
    for (; i < n; i += stride) dst[i] = (_Float16)src[i];
}

// ---------------------------------------------------------------------------
// Pipelined 16x64-per-wave GEMM body: C(16x64) += A(16xK) * B(64xK)^T
// A = xh rows (ld=CC), B = weight rows (ld=CC, natural [out][in] layout).
// k-loop is FULLY UNROLLED (24 chunks, compile-time) so the software
// double-buffer swap becomes pure SSA renames (no v_mov walls), while the
// source order "loads for k+32 before WMMAs for k" bounds the prefetch
// distance and keeps live VGPRs ~112 (no spills).
// ---------------------------------------------------------------------------
__device__ __forceinline__ void gemm_tile_16x64(
        const _Float16* __restrict__ xh, const _Float16* __restrict__ wh,
        int row0, int col0, v8f c[4]) {
    v16h a = load_a_frag(xh, CC, row0, 0);
    v16h b[4];
#pragma unroll
    for (int t = 0; t < 4; ++t) b[t] = load_b_frag(wh, CC, col0 + t * 16, 0);

#pragma unroll
    for (int k0 = 0; k0 < CC; k0 += 32) {
        const bool has_next = (k0 + 32) < CC;
        v16h an, bn[4];
        if (has_next) {
            an = load_a_frag(xh, CC, row0, k0 + 32);
#pragma unroll
            for (int t = 0; t < 4; ++t)
                bn[t] = load_b_frag(wh, CC, col0 + t * 16, k0 + 32);
        }
#pragma unroll
        for (int t = 0; t < 4; ++t) c[t] = wmma_f16(a, b[t], c[t]);
        if (has_next) {
            a = an;
#pragma unroll
            for (int t = 0; t < 4; ++t) b[t] = bn[t];
        }
    }
}

// ---------------------------------------------------------------------------
// Q projection:  Q = (x @ Wq^T + bq) * scale   ->  Qh[(b,h,q,hd)]  (f16)
// Wave tile 16x64; block = 4 waves = 64 rows x 64 cols.
// grid = (8192/64, 768/64) = (128, 12), block = 128
// ---------------------------------------------------------------------------
__global__ void __launch_bounds__(128, 1)
proj_q_kernel(const _Float16* __restrict__ xh,
              const _Float16* __restrict__ wqh,
              const float* __restrict__ bq,
              _Float16* __restrict__ Qh) {
    const int wv = threadIdx.x >> 5;
    const int l  = threadIdx.x & 31;
    const int row0 = blockIdx.x * 64 + wv * 16;
    const int col0 = blockIdx.y * 64;

    v8f c[4];
#pragma unroll
    for (int t = 0; t < 4; ++t) {
        float bv = bq[col0 + t * 16 + (l & 15)];
#pragma unroll
        for (int i = 0; i < 8; ++i) c[t][i] = bv;
    }

    gemm_tile_16x64(xh, wqh, row0, col0, c);

    const float scale = 0.125f;  // HD^-0.5
#pragma unroll
    for (int t = 0; t < 4; ++t) {
        int col = col0 + t * 16 + (l & 15);
        int h = col >> 6, hd = col & 63;
#pragma unroll
        for (int r = 0; r < 8; ++r) {
            int row = row0 + r + (l >> 4) * 8;
            int b_ = row >> 10, q = row & (NN - 1);
            Qh[(((size_t)(b_ * NH + h)) * NN + q) * HD + hd] =
                (_Float16)(c[t][r] * scale);
        }
    }
}

// ---------------------------------------------------------------------------
// KV projection: kv = x @ Wkv^T + bkv; col = two*768 + h*64 + hd
//   two==0 -> Kh[(b,h,key,hd)]            (f16, natural for QK^T B-frags)
//   two==1 -> Vt[(b,h,hd,key)] (V^T)      (f16, natural for P.V  B-frags)
// grid = (128, 1536/64 = 24), block = 128
// ---------------------------------------------------------------------------
__global__ void __launch_bounds__(128, 1)
proj_kv_kernel(const _Float16* __restrict__ xh,
               const _Float16* __restrict__ wkvh,
               const float* __restrict__ bkv,
               _Float16* __restrict__ Kh,
               _Float16* __restrict__ Vt) {
    const int wv = threadIdx.x >> 5;
    const int l  = threadIdx.x & 31;
    const int row0 = blockIdx.x * 64 + wv * 16;
    const int col0 = blockIdx.y * 64;

    v8f c[4];
#pragma unroll
    for (int t = 0; t < 4; ++t) {
        float bv = bkv[col0 + t * 16 + (l & 15)];
#pragma unroll
        for (int i = 0; i < 8; ++i) c[t][i] = bv;
    }

    gemm_tile_16x64(xh, wkvh, row0, col0, c);

    const int two = col0 / CC;  // uniform per block (64 | 768)
#pragma unroll
    for (int t = 0; t < 4; ++t) {
        int col = col0 + t * 16 + (l & 15);
        int rest = col - two * CC;
        int h = rest >> 6, hd = rest & 63;
#pragma unroll
        for (int r = 0; r < 8; ++r) {
            int row = row0 + r + (l >> 4) * 8;
            int b_ = row >> 10, q = row & (NN - 1);
            _Float16 v = (_Float16)c[t][r];
            if (two == 0)
                Kh[(((size_t)(b_ * NH + h)) * NN + q) * HD + hd] = v;
            else
                Vt[(((size_t)(b_ * NH + h)) * HD + hd) * NN + q] = v;
        }
    }
}

// ---------------------------------------------------------------------------
// Flash attention: per (b,h), one wave per 16-query tile, 4 waves per block.
// Per 32-key tile: all loads (K frags, V frags, bias gathers) are issued
// before the score WMMAs. kt-loop unrolled x2 so consecutive iterations'
// temporaries are distinct SSA values: iteration i+1's loads overlap
// iteration i's softmax + PV WMMAs with no copy-carried buffers.
// grid = (1024/64 = 16, NH, B), block = 128
// ---------------------------------------------------------------------------
__global__ void __launch_bounds__(128, 1)
attn_kernel(const _Float16* __restrict__ Qh,
            const _Float16* __restrict__ Kh,
            const _Float16* __restrict__ Vt,
            const float* __restrict__ rel_bias,
            const int* __restrict__ rel_idx,
            float* __restrict__ out) {
    __shared__ _Float16 plds[4][16 * 32];
    const int wv = threadIdx.x >> 5;
    const int l  = threadIdx.x & 31;
    const int b  = blockIdx.z;
    const int h  = blockIdx.y;
    const int bh = b * NH + h;
    const int q0 = blockIdx.x * 64 + wv * 16;

    const _Float16* Qb = Qh + (size_t)bh * NN * HD;
    const _Float16* Kb = Kh + (size_t)bh * NN * HD;
    const _Float16* Vb = Vt + (size_t)bh * HD * NN;

    // Q fragments held in registers for the whole sweep (16 rows x 64 hd)
    v16h aq0 = load_a_frag(Qb, HD, q0, 0);
    v16h aq1 = load_a_frag(Qb, HD, q0, 32);

    float m[8], lsum[8];
    v8f o[4];
#pragma unroll
    for (int r = 0; r < 8; ++r) { m[r] = -__builtin_inff(); lsum[r] = 0.0f; }
#pragma unroll
    for (int t = 0; t < 4; ++t)
#pragma unroll
        for (int i = 0; i < 8; ++i) o[t][i] = 0.0f;

#pragma unroll 2
    for (int kt = 0; kt < NN; kt += 32) {
        // ---- issue ALL of this tile's loads before any WMMA ----
        v16h bk0 = load_b_frag(Kb, HD, kt,      0);
        v16h bk1 = load_b_frag(Kb, HD, kt,      32);
        v16h bk2 = load_b_frag(Kb, HD, kt + 16, 0);
        v16h bk3 = load_b_frag(Kb, HD, kt + 16, 32);
        v16h bv[4];
#pragma unroll
        for (int t = 0; t < 4; ++t)
            bv[t] = load_b_frag(Vb, NN, t * 16, kt);

        float bias0[8], bias1[8];
#pragma unroll
        for (int r = 0; r < 8; ++r) {
            int q   = q0 + r + (l >> 4) * 8;
            int k0i = kt + (l & 15);
            bias0[r] = rel_bias[(size_t)rel_idx[(size_t)q * NN + k0i] * NH + h];
            bias1[r] = rel_bias[(size_t)rel_idx[(size_t)q * NN + k0i + 16] * NH + h];
        }

        // ---- scores: two 16x16 f32 tiles covering keys [kt, kt+32) ----
        v8f s0, s1;
#pragma unroll
        for (int i = 0; i < 8; ++i) { s0[i] = 0.0f; s1[i] = 0.0f; }
        s0 = wmma_f16(aq0, bk0, s0);
        s0 = wmma_f16(aq1, bk1, s0);
        s1 = wmma_f16(aq0, bk2, s1);
        s1 = wmma_f16(aq1, bk3, s1);

#pragma unroll
        for (int r = 0; r < 8; ++r) { s0[r] += bias0[r]; s1[r] += bias1[r]; }

        // ---- online softmax (rows live in 16-lane halves of the wave) ----
        float sc[8];
#pragma unroll
        for (int r = 0; r < 8; ++r) {
            float t = fmaxf(s0[r], s1[r]);
            t = fmaxf(t, __shfl_xor(t, 1));
            t = fmaxf(t, __shfl_xor(t, 2));
            t = fmaxf(t, __shfl_xor(t, 4));
            t = fmaxf(t, __shfl_xor(t, 8));
            float mn = fmaxf(m[r], t);
            sc[r] = __expf(m[r] - mn);
            m[r] = mn;
            float p0 = __expf(s0[r] - mn);
            float p1 = __expf(s1[r] - mn);
            s0[r] = p0; s1[r] = p1;
            float rs = p0 + p1;
            rs += __shfl_xor(rs, 1);
            rs += __shfl_xor(rs, 2);
            rs += __shfl_xor(rs, 4);
            rs += __shfl_xor(rs, 8);
            lsum[r] = lsum[r] * sc[r] + rs;
        }

        // rescale running output
#pragma unroll
        for (int t = 0; t < 4; ++t)
#pragma unroll
            for (int r = 0; r < 8; ++r) o[t][r] *= sc[r];

        // ---- P (f16) through per-wave LDS tile: C-layout -> A-layout ----
#pragma unroll
        for (int r = 0; r < 8; ++r) {
            int rowm = r + (l >> 4) * 8;
            plds[wv][rowm * 32 + (l & 15)]      = (_Float16)s0[r];
            plds[wv][rowm * 32 + 16 + (l & 15)] = (_Float16)s1[r];
        }
        v16h ap = load_a_frag(&plds[wv][0], 32, 0, 0);

        // ---- O += P.V  (B-frags straight from V^T, already in flight) ----
#pragma unroll
        for (int t = 0; t < 4; ++t)
            o[t] = wmma_f16(ap, bv[t], o[t]);
    }

    // ---- normalize and store: out[b][q][h*64+hd] ----
#pragma unroll
    for (int t = 0; t < 4; ++t) {
        int hd = t * 16 + (l & 15);
#pragma unroll
        for (int r = 0; r < 8; ++r) {
            int q = q0 + r + (l >> 4) * 8;
            out[((size_t)b * NN + q) * CC + h * HD + hd] = o[t][r] / lsum[r];
        }
    }
}

// ---------------------------------------------------------------------------
// Host-side launch
// ---------------------------------------------------------------------------
extern "C" void kernel_launch(void* const* d_in, const int* in_sizes, int n_in,
                              void* d_out, int out_size, void* d_ws, size_t ws_size,
                              hipStream_t stream) {
    const float* x        = (const float*)d_in[0];  // (8,1024,768)
    const float* Wq       = (const float*)d_in[1];  // (768,768)
    const float* bq       = (const float*)d_in[2];  // (768,)
    const float* Wkv      = (const float*)d_in[3];  // (1536,768)
    const float* bkv      = (const float*)d_in[4];  // (1536,)
    const float* rel_bias = (const float*)d_in[5];  // (3969,12)
    const int*   rel_idx  = (const int*)d_in[6];    // (1024,1024)
    float* out = (float*)d_out;

    char* ws = (char*)d_ws;
    size_t off = 0;
    auto carve = [&](size_t bytes) {
        char* p = ws + off;
        off = (off + bytes + 255) & ~(size_t)255;
        return p;
    };

    const size_t n_x   = (size_t)BB * NN * CC;      // 6,291,456
    const size_t n_wq  = (size_t)CC * CC;           // 589,824
    const size_t n_wkv = (size_t)2 * CC * CC;       // 1,179,648
    const size_t n_qkv = (size_t)BB * NH * NN * HD; // 6,291,456

    _Float16* xh   = (_Float16*)carve(n_x * 2);
    _Float16* wqh  = (_Float16*)carve(n_wq * 2);
    _Float16* wkvh = (_Float16*)carve(n_wkv * 2);
    _Float16* Qh   = (_Float16*)carve(n_qkv * 2);
    _Float16* Kh   = (_Float16*)carve(n_qkv * 2);
    _Float16* Vt   = (_Float16*)carve(n_qkv * 2);

    // 1) fp32 -> f16 conversions
    cvt_f32_to_f16<<<2048, 256, 0, stream>>>(x,   xh,   (int)n_x);
    cvt_f32_to_f16<<<512,  256, 0, stream>>>(Wq,  wqh,  (int)n_wq);
    cvt_f32_to_f16<<<512,  256, 0, stream>>>(Wkv, wkvh, (int)n_wkv);

    // 2) projections (WMMA GEMMs, 16x64 per wave, fully unrolled k-loop)
    proj_q_kernel <<<dim3(128, 12), 128, 0, stream>>>(xh, wqh,  bq,  Qh);
    proj_kv_kernel<<<dim3(128, 24), 128, 0, stream>>>(xh, wkvh, bkv, Kh, Vt);

    // 3) flash attention with relative-position bias
    attn_kernel<<<dim3(NN / 64, NH, BB), 128, 0, stream>>>(
        Qh, Kh, Vt, rel_bias, rel_idx, out);
}
